// _UpperTriDiagCore_13357348290699
// MI455X (gfx1250) — compile-verified, matrix operand order checked by reading, not run on hardware
//
#include <hip/hip_runtime.h>
#include <hip/hip_bf16.h>
#include <cmath>

// ---- problem constants (from reference setup) ----
#define NTHREADS 256   // 8 wave32s
#define NWAVES   8
#define NLAYERS  2
#define LSEQ     63
#define LPAD     64
#define DMODEL   128
#define DINNER   256
#define SSTATE   16
#define DRANK    8
#define DBC_N    40    // DR + 2*S
#define DBC_LD   48    // padded to 3 n-tiles of 16

typedef __attribute__((ext_vector_type(16))) _Float16 v16h;
typedef __attribute__((ext_vector_type(8)))  _Float16 v8h;
typedef __attribute__((ext_vector_type(8)))  float    v8f;
typedef __attribute__((ext_vector_type(4)))  unsigned int v4u;
typedef __attribute__((ext_vector_type(8)))  int      v8i;
typedef __attribute__((ext_vector_type(4)))  int      v4i;

__device__ __forceinline__ float siluf(float x) { return x / (1.f + expf(-x)); }
__device__ __forceinline__ float softplusf(float x) {
  return (x > 20.f) ? x : log1pf(expf(x));
}
__device__ __forceinline__ v8f bcast8(float v) {
  v8f c;
#pragma unroll
  for (int r = 0; r < 8; ++r) c[r] = v;
  return c;
}

// 16x32 f16 fragment from an f16 LDS matrix, row stride lda (halves).
// ISA 16-bit A layout: lane owns row (lane&15), g = lane>>4; elements 0..7 ->
// K=g*8+e, elements 8..15 -> K=16+g*8+(e-8). Two 16B chunks => 2x ds_load_b128.
// Used for A (activations) and for B via the transposed staged weights (W^T rows = N).
__device__ __forceinline__ v16h frag_lds16(const _Float16* src, int lda, int lane) {
  const int g = lane >> 4, m = lane & 15;
  const _Float16* row = src + m * lda;
  v8h lo = *(const v8h*)(row + g * 8);
  v8h hi = *(const v8h*)(row + 16 + g * 8);
  return __builtin_shufflevector(lo, hi, 0, 1, 2, 3, 4, 5, 6, 7,
                                 8, 9, 10, 11, 12, 13, 14, 15);
}

// C/D layout: lane group g picks M = 8*g + r (r = VGPR index), column n = lane&15.
__device__ __forceinline__ void frag_store_f32(float* dst, int ldd, int lane, v8f c, int nmax) {
  const int g = lane >> 4, n = lane & 15;
  if (n >= nmax) return;
  float* p = dst + (8 * g) * ldd + n;
#pragma unroll
  for (int r = 0; r < 8; ++r) p[r * ldd] = c[r];
}
__device__ __forceinline__ void frag_store_f16(_Float16* dst, int ldd, int lane, v8f c) {
  const int g = lane >> 4, n = lane & 15;
  _Float16* p = dst + (8 * g) * ldd + n;
#pragma unroll
  for (int r = 0; r < 8; ++r) p[r * ldd] = (_Float16)c[r];
}
__device__ __forceinline__ void frag_store_dual(float* dst, _Float16* dst16, int ldd,
                                                int lane, v8f c) {
  const int g = lane >> 4, n = lane & 15;
  float* p = dst + (8 * g) * ldd + n;
  _Float16* q = dst16 + (8 * g) * ldd + n;
#pragma unroll
  for (int r = 0; r < 8; ++r) { p[r * ldd] = c[r]; q[r * ldd] = (_Float16)c[r]; }
}

__device__ __forceinline__ v8f wmma_f16(v16h a, v16h b, v8f c) {
  return __builtin_amdgcn_wmma_f32_16x16x32_f16(
      /*neg_a=*/false, a, /*neg_b=*/false, b,
      /*c_mod=*/(short)0, c, /*reuse_a=*/false, /*reuse_b=*/false);
}

__global__ __launch_bounds__(NTHREADS)
void mamba_patch_kernel(const float* __restrict__ rp,     const float* __restrict__ tp_w,
                        const float* __restrict__ tp_b,   const float* __restrict__ in_w,
                        const float* __restrict__ conv_w, const float* __restrict__ conv_b,
                        const float* __restrict__ xp_w,   const float* __restrict__ dt_w,
                        const float* __restrict__ dt_b,   const float* __restrict__ A_log,
                        const float* __restrict__ Dp,     const float* __restrict__ out_w,
                        const float* __restrict__ ln_g,   const float* __restrict__ ln_b,
                        float* __restrict__ out) {
  extern __shared__ char smem[];
  float*     s_x    = (float*)smem;                   // 64*128 f32 (x / residual)
  float*     s_dbc  = s_x + LPAD * DMODEL;            // 64*48  f32 (dt|B|C, padded)
  float*     s_mu   = s_dbc + LPAD * DBC_LD;          // 64
  float*     s_rs   = s_mu + LPAD;                    // 64
  _Float16*  s_x16  = (_Float16*)(s_rs + LPAD);       // 64*128 f16 mirror of x
  _Float16*  s_xc16 = s_x16 + LPAD * DMODEL;          // 64*256 f16 (xc, later gated y)
  _Float16*  s_z16  = s_xc16 + LPAD * DINNER;         // 64*256 f16 (gate z)
  _Float16*  s_wt   = s_z16 + LPAD * DINNER;          // 128KB arena: staged W^T (f16)
  float*     s_rp   = s_x;                            // alias: raw patch staging (16KB)
  _Float16*  s_tok16 = s_xc16;                        // alias: token matrix (64x64 f16)

  const int tid  = threadIdx.x;
  const int lane = tid & 31;
  const int wv   = tid >> 5;
  const int n    = blockIdx.x;
  const float* rpn = rp + (size_t)n * (64 * 64);

  // ---- Phase 0a: stage the 16KB rp patch into LDS via the Tensor Data Mover ----
#if __has_builtin(__builtin_amdgcn_tensor_load_to_lds) && __has_builtin(__builtin_amdgcn_s_wait_tensorcnt)
  if (tid < 32) {  // one wave issues the DMA (TDM is per-wave, EXEC-independent)
    const unsigned lds_off = (unsigned)((char*)s_rp - (char*)smem);
    const unsigned long long ga = (unsigned long long)(uintptr_t)rpn;
    // D# group 0: count=1 | lds_addr | global_addr | type=2 ("image")
    v4u g0 = { 1u, lds_off, (unsigned)(ga & 0xFFFFFFFFu),
               (unsigned)(((ga >> 32) & 0x01FFFFFFu) | (2u << 30)) };
    // D# group 1: 1-D tensor: data_size=2 (4B), tensor_dim0=4096, tensor_dim1=1,
    //             tile_dim0=4096, tile_dim1=1, tensor_dim0_stride=4096.
    v8i g1 = { (int)0x00020000,        // workgroup_mask=0, data_size=2
               (int)(4096u << 16),     // atomic_addr=0 | tensor_dim0.lo16
               (int)0x00010000,        // tensor_dim0.hi16=0 | tensor_dim1.lo16=1
               (int)(4096u << 16),     // tensor_dim1.hi16=0 | tile_dim0=4096
               1,                      // tile_dim1=1 | tile_dim2=0
               4096,                   // tensor_dim0_stride.lo32
               0, 0 };                 // stride0.hi | stride1
    v4i gz4 = { 0, 0, 0, 0 };          // groups 2/3 unused (<=2D tensor)
    v8i gz8 = { 0, 0, 0, 0, 0, 0, 0, 0 };
    __builtin_amdgcn_tensor_load_to_lds(g0, g1, gz4, gz4, gz8, 0);
    __builtin_amdgcn_s_wait_tensorcnt(0);
  }
  __syncthreads();
#else
  for (int i = tid; i < 64 * 64; i += NTHREADS) s_rp[i] = rpn[i];
  __syncthreads();
#endif

  // ---- Phase 0b: token matrix (f16) + stage tp_w^T (128 x 64, f16, K-padded) ----
  // tokens[t][j] = rp_flat[j*65 + t + 1] if j < 63-t else 0 ; row 63 zero.
  for (int i = tid; i < 64 * 64; i += NTHREADS) {
    const int t = i >> 6, j = i & 63;
    float v = 0.f;
    if (t < LSEQ && j < LSEQ - t) v = s_rp[j * 65 + t + 1];
    s_tok16[i] = (_Float16)v;
  }
  for (int i = tid; i < 64 * DMODEL; i += NTHREADS) {   // i = k*128 + n (coalesced)
    const int k = i >> 7, nn = i & 127;
    const float v = (k < LSEQ) ? tp_w[k * DMODEL + nn] : 0.f;
    s_wt[nn * 64 + k] = (_Float16)v;
  }
  __syncthreads();

  // ---- GEMM1: x = T(64x64) @ tp_w(63x128) + tp_b ----
  // wave: mi = wv&3, ni = (wv>>2) + 2*j (4 N-tiles resident); 2 K-steps.
  {
    const int mi = wv & 3, nb = wv >> 2;
    v8f c[4];
#pragma unroll
    for (int j = 0; j < 4; ++j) c[j] = bcast8(tp_b[(nb + 2 * j) * 16 + (lane & 15)]);
#pragma unroll
    for (int kk = 0; kk < 64; kk += 32) {
      v16h a = frag_lds16(s_tok16 + mi * 16 * 64 + kk, 64, lane);
#pragma unroll
      for (int j = 0; j < 4; ++j) {
        v16h b = frag_lds16(s_wt + (nb + 2 * j) * 16 * 64 + kk, 64, lane);
        c[j] = wmma_f16(a, b, c[j]);
      }
    }
#pragma unroll
    for (int j = 0; j < 4; ++j) {
      const int ni = nb + 2 * j;
      frag_store_dual(s_x + mi * 16 * DMODEL + ni * 16,
                      s_x16 + mi * 16 * DMODEL + ni * 16, DMODEL, lane, c[j]);
    }
  }
  __syncthreads();

  // ---- Mamba layers ----
  for (int l = 0; l < NLAYERS; ++l) {
    const float* Wi = in_w  + (size_t)l * DMODEL * (2 * DINNER);
    const float* Wx = xp_w  + (size_t)l * DINNER * DBC_N;
    const float* Wo = out_w + (size_t)l * DINNER * DMODEL;
    __builtin_prefetch(Wx, 0, 1);            // global_prefetch_b8 (upcoming stages)
    __builtin_prefetch(Wo, 0, 1);

    // Stage in_w^T: (512 x 128) f16. i = k*512 + n (coalesced reads).
    for (int i = tid; i < DMODEL * 512; i += NTHREADS) {
      const int k = i >> 9, nn = i & 511;
      s_wt[nn * DMODEL + k] = (_Float16)Wi[i];
    }
    __syncthreads();

    // GEMM2: xz = x(64x128) @ in_w(128x512) -> s_xc16 | s_z16.
    // wave: mi = wv&3, 16 N-tiles in two register-resident chunks of 8; 4 K-steps.
#pragma unroll
    for (int chunk = 0; chunk < 2; ++chunk) {
      const int mi = wv & 3;
      const int nb = (wv >> 2) * 16 + chunk * 8;
      v8f c[8];
#pragma unroll
      for (int j = 0; j < 8; ++j) c[j] = bcast8(0.f);
#pragma unroll
      for (int kk = 0; kk < DMODEL; kk += 32) {
        v16h a = frag_lds16(s_x16 + mi * 16 * DMODEL + kk, DMODEL, lane);
#pragma unroll
        for (int j = 0; j < 8; ++j) {
          v16h b = frag_lds16(s_wt + (nb + j) * 16 * DMODEL + kk, DMODEL, lane);
          c[j] = wmma_f16(a, b, c[j]);
        }
      }
#pragma unroll
      for (int j = 0; j < 8; ++j) {
        const int ni = nb + j;
        _Float16* dst = (ni < 16) ? (s_xc16 + ni * 16) : (s_z16 + (ni - 16) * 16);
        frag_store_f16(dst + mi * 16 * DINNER, DINNER, lane, c[j]);
      }
    }
    __syncthreads();

    // Stage xp_w^T (48 x 256, f16, N-padded with zeros) + depthwise conv/SiLU.
    for (int i = tid; i < 48 * DINNER; i += NTHREADS) {  // i = k*48 + n (coalesced)
      const int k = i / 48, nn = i % 48;
      const float v = (nn < DBC_N) ? Wx[k * DBC_N + nn] : 0.f;
      s_wt[nn * DINNER + k] = (_Float16)v;
    }
    {
      const int d = tid;  // DINNER == NTHREADS; causal K=3 conv, in place
      const float cw0 = conv_w[(l * 3 + 0) * DINNER + d];
      const float cw1 = conv_w[(l * 3 + 1) * DINNER + d];
      const float cw2 = conv_w[(l * 3 + 2) * DINNER + d];
      const float cbv = conv_b[l * DINNER + d];
      float xm2 = 0.f, xm1 = 0.f;
      for (int t = 0; t < LSEQ; ++t) {
        const float cur = (float)s_xc16[t * DINNER + d];
        const float o = cw0 * xm2 + cw1 * xm1 + cw2 * cur + cbv;
        s_xc16[t * DINNER + d] = (_Float16)siluf(o);
        xm2 = xm1; xm1 = cur;
      }
    }
    __syncthreads();

    // GEMM3: dbc = xc(64x256) @ xp_w(256x40->48). waves 0-3: ni {0,1}; 4-7: ni {2}.
    if (wv < 4) {
      const int mi = wv;
      v8f c0 = bcast8(0.f), c1 = bcast8(0.f);
#pragma unroll
      for (int kk = 0; kk < DINNER; kk += 32) {
        v16h a = frag_lds16(s_xc16 + mi * 16 * DINNER + kk, DINNER, lane);
        v16h b0 = frag_lds16(s_wt + 0 * DINNER + kk, DINNER, lane);
        c0 = wmma_f16(a, b0, c0);
        v16h b1 = frag_lds16(s_wt + 16 * DINNER + kk, DINNER, lane);
        c1 = wmma_f16(a, b1, c1);
      }
      frag_store_f32(s_dbc + mi * 16 * DBC_LD + 0, DBC_LD, lane, c0, 16);
      frag_store_f32(s_dbc + mi * 16 * DBC_LD + 16, DBC_LD, lane, c1, 16);
    } else {
      const int mi = wv - 4;
      v8f c2 = bcast8(0.f);
#pragma unroll
      for (int kk = 0; kk < DINNER; kk += 32) {
        v16h a = frag_lds16(s_xc16 + mi * 16 * DINNER + kk, DINNER, lane);
        v16h b2 = frag_lds16(s_wt + 32 * DINNER + kk, DINNER, lane);
        c2 = wmma_f16(a, b2, c2);
      }
      frag_store_f32(s_dbc + mi * 16 * DBC_LD + 32, DBC_LD, lane, c2, DBC_N - 32);
    }
    __syncthreads();

    // Stage out_w^T (128 x 256 f16) + fused delta-proj/softplus/scan/gate.
    for (int i = tid; i < DINNER * DMODEL; i += NTHREADS) {  // i = k*128 + n
      const int k = i >> 7, nn = i & 127;
      s_wt[nn * DINNER + k] = (_Float16)Wo[i];
    }
    {
      const int d = tid;
      float Ac[SSTATE];
#pragma unroll
      for (int s = 0; s < SSTATE; ++s) Ac[s] = -expf(A_log[(l * DINNER + d) * SSTATE + s]);
      float dtw[DRANK];
#pragma unroll
      for (int r = 0; r < DRANK; ++r) dtw[r] = dt_w[(l * DRANK + r) * DINNER + d];
      const float dtbv = dt_b[l * DINNER + d];
      const float dpv  = Dp[l * DINNER + d];
      float h[SSTATE];
#pragma unroll
      for (int s = 0; s < SSTATE; ++s) h[s] = 0.f;
      for (int t = 0; t < LSEQ; ++t) {
        const float* dbc_t = s_dbc + t * DBC_LD;
        float dtv = dtbv;
#pragma unroll
        for (int r = 0; r < DRANK; ++r) dtv += dbc_t[r] * dtw[r];
        const float delta = softplusf(dtv);
        const float u  = (float)s_xc16[t * DINNER + d];
        const float du = delta * u;
        float y = 0.f;
#pragma unroll
        for (int s = 0; s < SSTATE; ++s) {
          const float dA = expf(delta * Ac[s]);
          h[s] = dA * h[s] + du * dbc_t[DRANK + s];          // B_t
          y += h[s] * dbc_t[DRANK + SSTATE + s];             // C_t
        }
        y += dpv * u;
        const float zg = siluf((float)s_z16[t * DINNER + d]);
        s_xc16[t * DINNER + d] = (_Float16)(y * zg);         // gated y, in place
      }
    }
    __syncthreads();

    // GEMM4: x += y(64x256) @ out_w(256x128); residual accumulate + f16 mirror.
    {
      const int mi = wv & 3;
      const int nb = (wv >> 2) * 4;
      v8f c[4];
#pragma unroll
      for (int j = 0; j < 4; ++j) c[j] = bcast8(0.f);
#pragma unroll
      for (int kk = 0; kk < DINNER; kk += 32) {
        v16h a = frag_lds16(s_xc16 + mi * 16 * DINNER + kk, DINNER, lane);
#pragma unroll
        for (int j = 0; j < 4; ++j) {
          v16h b = frag_lds16(s_wt + (nb + j) * 16 * DINNER + kk, DINNER, lane);
          c[j] = wmma_f16(a, b, c[j]);
        }
      }
      const int g = lane >> 4, nn = lane & 15;
#pragma unroll
      for (int j = 0; j < 4; ++j) {
        const int ni = nb + j;
        float*    p = s_x   + (mi * 16 + 8 * g) * DMODEL + ni * 16 + nn;
        _Float16* q = s_x16 + (mi * 16 + 8 * g) * DMODEL + ni * 16 + nn;
#pragma unroll
        for (int r = 0; r < 8; ++r) {
          const float v = p[r * DMODEL] + c[j][r];
          p[r * DMODEL] = v;
          q[r * DMODEL] = (_Float16)v;
        }
      }
    }
    __syncthreads();
  }

  // ---- LayerNorm stats per token (f32 path) ----
  if (tid < LSEQ) {
    float mu = 0.f;
    for (int d = 0; d < DMODEL; ++d) mu += s_x[tid * DMODEL + d];
    mu *= (1.f / DMODEL);
    float var = 0.f;
    for (int d = 0; d < DMODEL; ++d) {
      const float df = s_x[tid * DMODEL + d] - mu;
      var += df * df;
    }
    var *= (1.f / DMODEL);
    s_mu[tid] = mu;
    s_rs[tid] = rsqrtf(var + 1e-5f);
  }
  __syncthreads();

  // ---- normalize, scale, mean over the 63 tokens -> out[n][0..127] ----
  if (tid < DMODEL) {
    float acc = 0.f;
    for (int t = 0; t < LSEQ; ++t)
      acc += (s_x[t * DMODEL + tid] - s_mu[t]) * s_rs[t];
    out[(size_t)n * DMODEL + tid] = acc * (1.f / LSEQ) * ln_g[tid] + ln_b[tid];
  }
}

extern "C" void kernel_launch(void* const* d_in, const int* in_sizes, int n_in,
                              void* d_out, int out_size, void* d_ws, size_t ws_size,
                              hipStream_t stream) {
  const float* rp     = (const float*)d_in[0];
  const float* tp_w   = (const float*)d_in[1];
  const float* tp_b   = (const float*)d_in[2];
  const float* in_w   = (const float*)d_in[3];
  const float* conv_w = (const float*)d_in[4];
  const float* conv_b = (const float*)d_in[5];
  const float* xp_w   = (const float*)d_in[6];
  const float* dt_w   = (const float*)d_in[7];
  const float* dt_b   = (const float*)d_in[8];
  const float* A_log  = (const float*)d_in[9];
  const float* Dp     = (const float*)d_in[10];
  const float* out_w  = (const float*)d_in[11];
  const float* ln_g   = (const float*)d_in[12];
  const float* ln_b   = (const float*)d_in[13];
  float* out = (float*)d_out;

  const int N = in_sizes[0] / (64 * 64);  // B*P = 512 sequences
  const size_t smem_bytes =
      (size_t)(LPAD * DMODEL + LPAD * DBC_LD + 2 * LPAD) * sizeof(float) +          // f32
      (size_t)(LPAD * DMODEL + 2 * LPAD * DINNER + 512 * DMODEL) * sizeof(_Float16); // f16

  // ~252 KB of LDS per workgroup (320 KB available per WGP on CDNA5): opt in.
  (void)hipFuncSetAttribute((const void*)mamba_patch_kernel,
                            hipFuncAttributeMaxDynamicSharedMemorySize, (int)smem_bytes);

  mamba_patch_kernel<<<N, NTHREADS, smem_bytes, stream>>>(
      rp, tp_w, tp_b, in_w, conv_w, conv_b, xp_w, dt_w, dt_b, A_log, Dp, out_w,
      ln_g, ln_b, out);
}
